// UVSeamGNN_65231963292249
// MI455X (gfx1250) — matrix operand order repaired and checked by reading, not compiled, not run on hardware
//
#include <hip/hip_runtime.h>

// ---------------------------------------------------------------------------
// UVSeamGNN for MI455X (gfx1250, wave32, WMMA bf16 16x16x32)
// ---------------------------------------------------------------------------

typedef __attribute__((ext_vector_type(16))) __bf16 v16bf;
typedef __attribute__((ext_vector_type(8)))  float  v8f;

#define HDIM 128
#define DIN  6
#define EIN  11

// ---------------------------------------------------------------------------
// WMMA helpers (CDNA5 V_WMMA_F32_16X16X32_BF16)
// A fragment: 16x32 (MxK) bf16. lane L: row M = L&15 ; kbase = (L>>4)*8
//   elements 0..7  -> K = kbase + 0..7
//   elements 8..15 -> K = 16 + kbase + 0..7
// B fragment: 32x16 (KxN) bf16, sourced from row-major W[n][k] (B[k][n]=W[n][k]).
//   lane L: col N = L&15 ; same K mapping as A fragment
// C/D: element r -> M = r + 8*(lane>>4), N = lane&15
// ---------------------------------------------------------------------------
__device__ __forceinline__ v16bf load_a_frag(const __bf16* M, int ld, int row0,
                                             int k0, int lane) {
  const __bf16* p = M + (size_t)(row0 + (lane & 15)) * ld + k0 + ((lane >> 4) << 3);
  v16bf a;
#pragma unroll
  for (int e = 0; e < 8; ++e) a[e] = p[e];
#pragma unroll
  for (int e = 0; e < 8; ++e) a[8 + e] = p[16 + e];
  return a;
}

__device__ __forceinline__ v16bf load_b_frag(const __bf16* W, int ld, int n0,
                                             int k0, int lane) {
  const __bf16* p = W + (size_t)(n0 + (lane & 15)) * ld + k0 + ((lane >> 4) << 3);
  v16bf b;
#pragma unroll
  for (int e = 0; e < 8; ++e) b[e] = p[e];
#pragma unroll
  for (int e = 0; e < 8; ++e) b[8 + e] = p[16 + e];
  return b;
}

__device__ __forceinline__ v8f wmma_bf16(v16bf a, v16bf b, v8f c) {
  return __builtin_amdgcn_wmma_f32_16x16x32_bf16(false, a, false, b, (short)0, c,
                                                 false, false);
}

// ---------------------------------------------------------------------------
// Utility kernels
// ---------------------------------------------------------------------------
__global__ void zero_f32(float* p, long n) {
  long i = (long)blockIdx.x * blockDim.x + threadIdx.x;
  if (i < n) p[i] = 0.0f;
}

__global__ void cvt_bf16(const float* __restrict__ s, __bf16* __restrict__ d, long n) {
  long i = (long)blockIdx.x * blockDim.x + threadIdx.x;
  if (i < n) d[i] = (__bf16)s[i];
}

// Split We1 [128 x 267] into bf16 [128x128] (cols 0..127) and [128x128] (128..255)
__global__ void split_we1(const float* __restrict__ We1, __bf16* __restrict__ a,
                          __bf16* __restrict__ b) {
  int i = blockIdx.x * blockDim.x + threadIdx.x;
  if (i < HDIM * HDIM) {
    int j = i >> 7, k = i & 127;
    a[i] = (__bf16)We1[j * 267 + k];
    b[i] = (__bf16)We1[j * 267 + 128 + k];
  }
}

// ---------------------------------------------------------------------------
// Degree / aggregation
// ---------------------------------------------------------------------------
__global__ void deg_kernel(const int* __restrict__ dst, float* __restrict__ cnt, int E) {
  int e = blockIdx.x * blockDim.x + threadIdx.x;
  if (e < E) atomicAdd(&cnt[dst[e]], 1.0f);
}

__global__ void inv_kernel(const float* __restrict__ cnt, float* __restrict__ inv, int n) {
  int i = blockIdx.x * blockDim.x + threadIdx.x;
  if (i < n) inv[i] = 1.0f / fmaxf(cnt[i], 1.0f);
}

__global__ void agg6_kernel(const int* __restrict__ src, const int* __restrict__ dst,
                            const float* __restrict__ x, float* __restrict__ sum6, int E) {
  int e = blockIdx.x * blockDim.x + threadIdx.x;
  if (e < E) {
    int s = src[e], d = dst[e];
#pragma unroll
    for (int k = 0; k < DIN; ++k)
      atomicAdd(&sum6[(size_t)d * DIN + k], x[(size_t)s * DIN + k]);
  }
}

// one thread per (edge, 4-feature chunk): float4 load + 4 component atomics.
// accumulator table (51 MB) is L2-resident on the 192 MB L2.
__global__ void aggH_kernel(const int* __restrict__ src, const int* __restrict__ dst,
                            const float* __restrict__ h, float* __restrict__ sum, long EC) {
  long i = (long)blockIdx.x * blockDim.x + threadIdx.x;
  if (i < EC) {
    long e = i >> 5;
    int f = (int)(i & 31) << 2;
    int s = src[e], d = dst[e];
    float4 v = *(const float4*)(h + (size_t)s * HDIM + f);
    float* p = sum + (size_t)d * HDIM + f;
    atomicAdd(p + 0, v.x);
    atomicAdd(p + 1, v.y);
    atomicAdd(p + 2, v.z);
    atomicAdd(p + 3, v.w);
  }
}

__global__ void meanbf_kernel(const float* __restrict__ sum, const float* __restrict__ inv,
                              __bf16* __restrict__ mbf, long n) {
  long i = (long)blockIdx.x * blockDim.x + threadIdx.x;
  if (i < n) mbf[i] = (__bf16)(sum[i] * inv[i >> 7]);
}

// ---------------------------------------------------------------------------
// Layer 1 node transform (K = 6, scalar VALU)
// h1 = relu(mean @ W1l.T + b1l + x @ W1r.T)
// ---------------------------------------------------------------------------
__global__ void node1_kernel(const float* __restrict__ x, const float* __restrict__ sum6,
                             const float* __restrict__ inv, const float* __restrict__ W1l,
                             const float* __restrict__ b1l, const float* __restrict__ W1r,
                             float* __restrict__ h1, __bf16* __restrict__ h1bf, int n) {
  int node = blockIdx.x;
  int j = threadIdx.x;
  __shared__ float xs[DIN], ms[DIN];
  if (j < DIN) xs[j] = x[(size_t)node * DIN + j];
  else if (j < 2 * DIN) {
    int k = j - DIN;
    ms[k] = sum6[(size_t)node * DIN + k] * inv[node];
  }
  __syncthreads();
  float acc = b1l[j];
#pragma unroll
  for (int k = 0; k < DIN; ++k)
    acc += ms[k] * W1l[j * DIN + k] + xs[k] * W1r[j * DIN + k];
  acc = fmaxf(acc, 0.0f);
  h1[(size_t)node * HDIM + j] = acc;
  h1bf[(size_t)node * HDIM + j] = (__bf16)acc;
}

// ---------------------------------------------------------------------------
// Layers 2/3 node transform via WMMA:
// out = relu(meanbf @ Wl.T + bl + hbf @ Wr.T) [+ res]
// block = 256 thr (8 waves) -> 16 nodes x 128 out cols; wave w: cols w*16..w*16+15
// ---------------------------------------------------------------------------
__global__ void __launch_bounds__(256)
sage_node_wmma(const __bf16* __restrict__ meanbf, const __bf16* __restrict__ hbf,
               const __bf16* __restrict__ Wl, const float* __restrict__ bl,
               const __bf16* __restrict__ Wr, const float* __restrict__ res,
               float* __restrict__ out, __bf16* __restrict__ outbf, int n_nodes) {
  const int lane = threadIdx.x & 31;
  const int wave = threadIdx.x >> 5;
  const int row0 = blockIdx.x * 16;
  const int j0 = wave * 16;

  v8f c = {};
#pragma unroll
  for (int kc = 0; kc < HDIM; kc += 32) {
    v16bf a = load_a_frag(meanbf, HDIM, row0, kc, lane);
    v16bf b = load_b_frag(Wl, HDIM, j0, kc, lane);
    c = wmma_bf16(a, b, c);
    a = load_a_frag(hbf, HDIM, row0, kc, lane);
    b = load_b_frag(Wr, HDIM, j0, kc, lane);
    c = wmma_bf16(a, b, c);
  }

  const int j = j0 + (lane & 15);
  const float bias = bl[j];
#pragma unroll
  for (int r = 0; r < 8; ++r) {
    int m = r + ((lane >> 4) << 3);
    int node = row0 + m;
    if (node < n_nodes) {
      float v = fmaxf(c[r] + bias, 0.0f);
      if (res) v += res[(size_t)node * HDIM + j];
      out[(size_t)node * HDIM + j] = v;
      outbf[(size_t)node * HDIM + j] = (__bf16)v;
    }
  }
}

// ---------------------------------------------------------------------------
// Fused edge MLP: 64 edges per block (256 thr = 8 waves).
// Stage 1: wave w owns cols w*16..+15, loads B-frags ONCE per K-step and
//          reuses across 4 M-tiles (4x weight-traffic amortization):
//   z1 = relu(h3[src]@We1a.T + h3[dst]@We1b.T + ea@We1c.T + be1)   [64x128]
// Stage 2: 16 (M-tile, j-tile) jobs over 8 waves:
//   z2 = relu(z1 @ We2.T + be2)                                     [64x64]
// Stage 3: out = z2 @ We3.T + be3                                   [64]
// LDS reuse: z1 overwrites the As gather buffer; z2 overwrites Ad.
// ---------------------------------------------------------------------------
#define EPB 64    // edges per block
#define LDA 136   // bf16 stride (272 B): 16-row fragment reads hit distinct banks
#define LDZ2 68   // f32 stride (272 B): conflict-free; 64*68*4 == 64*136*2 bytes

__global__ void __launch_bounds__(256)
edge_mlp_kernel(const int* __restrict__ src, const int* __restrict__ dst,
                const float* __restrict__ edge_attr, const __bf16* __restrict__ h3bf,
                const __bf16* __restrict__ We1a, const __bf16* __restrict__ We1b,
                const float* __restrict__ We1, const float* __restrict__ be1,
                const __bf16* __restrict__ We2bf, const float* __restrict__ be2,
                const float* __restrict__ We3, const float* __restrict__ be3,
                float* __restrict__ out, int E) {
  __shared__ __align__(16) char smem[EPB * LDA * 2 * 2 + EPB * 12 * 4]; // 37,888 B
  __bf16* As  = (__bf16*)smem;                        // [EPB][LDA] gather src
  __bf16* Ad  = (__bf16*)(smem + EPB * LDA * 2);      // [EPB][LDA] gather dst
  float*  eas = (float*)(smem + EPB * LDA * 4);       // [EPB][12] edge attrs
  __bf16* z1s = As;                                   // reused after barrier
  float*  z2s = (float*)(smem + EPB * LDA * 2);       // reuses Ad region

  const int t = threadIdx.x;
  const int lane = t & 31;
  const int wave = t >> 5;
  const int e0 = blockIdx.x * EPB;

  // ---- gather h3 rows for src/dst into LDS (uint4 = 8 bf16 per copy) ----
#pragma unroll
  for (int it = 0; it < EPB / 16; ++it) {
    int m = it * 16 + (t >> 4), ch = t & 15;          // 64 rows x 16 chunks
    int e = min(e0 + m, E - 1);
    size_t rs = (size_t)src[e] * HDIM + ch * 8;
    size_t rd = (size_t)dst[e] * HDIM + ch * 8;
    *(uint4*)&As[m * LDA + ch * 8] = *(const uint4*)(h3bf + rs);
    *(uint4*)&Ad[m * LDA + ch * 8] = *(const uint4*)(h3bf + rd);
  }
  for (int i = t; i < EPB * EIN; i += 256) {
    int mm = i / EIN, kk = i % EIN;
    int ee = min(e0 + mm, E - 1);
    eas[mm * 12 + kk] = edge_attr[(size_t)ee * EIN + kk];
  }
  __syncthreads();

  // ---- stage 1: two dense GEMMs (K=128), B-frags reused over 4 M-tiles ----
  const int j0 = wave * 16;
  v8f c1[4] = {{}, {}, {}, {}};
#pragma unroll
  for (int kc = 0; kc < HDIM; kc += 32) {
    v16bf ba = load_b_frag(We1a, HDIM, j0, kc, lane);
    v16bf bb = load_b_frag(We1b, HDIM, j0, kc, lane);
#pragma unroll
    for (int mt = 0; mt < 4; ++mt) {
      v16bf a = load_a_frag(As, LDA, mt * 16, kc, lane);
      c1[mt] = wmma_bf16(a, ba, c1[mt]);
      a = load_a_frag(Ad, LDA, mt * 16, kc, lane);
      c1[mt] = wmma_bf16(a, bb, c1[mt]);
    }
  }
  // epilogue (bias + 11-wide edge-attr correction + relu), in registers
  {
    const int j = j0 + (lane & 15);
    float wc[EIN];
#pragma unroll
    for (int k = 0; k < EIN; ++k) wc[k] = We1[j * 267 + 256 + k];
    const float bias = be1[j];
#pragma unroll
    for (int mt = 0; mt < 4; ++mt)
#pragma unroll
      for (int r = 0; r < 8; ++r) {
        int m = mt * 16 + r + ((lane >> 4) << 3);
        float v = c1[mt][r] + bias;
#pragma unroll
        for (int k = 0; k < EIN; ++k) v += eas[m * 12 + k] * wc[k];
        c1[mt][r] = fmaxf(v, 0.0f);
      }
  }
  __syncthreads();                                    // done reading As/Ad
  {
    const int j = j0 + (lane & 15);
#pragma unroll
    for (int mt = 0; mt < 4; ++mt)
#pragma unroll
      for (int r = 0; r < 8; ++r) {
        int m = mt * 16 + r + ((lane >> 4) << 3);
        z1s[m * LDA + j] = (__bf16)c1[mt][r];
      }
  }
  __syncthreads();

  // ---- stage 2: z2 = relu(z1 @ We2.T + be2); 16 tile-jobs over 8 waves ----
#pragma unroll
  for (int i = 0; i < 2; ++i) {
    int job = wave * 2 + i;
    int row0 = (job >> 2) * 16;
    int jj = (job & 3) * 16;
    v8f c = {};
#pragma unroll
    for (int kc = 0; kc < HDIM; kc += 32) {
      v16bf a = load_a_frag(z1s, LDA, row0, kc, lane);
      v16bf b = load_b_frag(We2bf, HDIM, jj, kc, lane);
      c = wmma_bf16(a, b, c);
    }
    const int j = jj + (lane & 15);
    const float bias = be2[j];
#pragma unroll
    for (int r = 0; r < 8; ++r) {
      int m = row0 + r + ((lane >> 4) << 3);
      z2s[m * LDZ2 + j] = fmaxf(c[r] + bias, 0.0f);
    }
  }
  __syncthreads();

  // ---- stage 3: 64-wide dot per edge ----
  if (t < EPB && e0 + t < E) {
    float acc = be3[0];
#pragma unroll
    for (int k = 0; k < 64; ++k) acc += z2s[t * LDZ2 + k] * We3[k];
    out[e0 + t] = acc;
  }
}

// ---------------------------------------------------------------------------
// Host launch
// ---------------------------------------------------------------------------
extern "C" void kernel_launch(void* const* d_in, const int* in_sizes, int n_in,
                              void* d_out, int out_size, void* d_ws, size_t ws_size,
                              hipStream_t stream) {
  const float* x         = (const float*)d_in[0];
  const int*   eidx      = (const int*)d_in[1];
  const float* edge_attr = (const float*)d_in[2];
  const float* W1l = (const float*)d_in[3];
  const float* b1l = (const float*)d_in[4];
  const float* W1r = (const float*)d_in[5];
  const float* W2l = (const float*)d_in[6];
  const float* b2l = (const float*)d_in[7];
  const float* W2r = (const float*)d_in[8];
  const float* W3l = (const float*)d_in[9];
  const float* b3l = (const float*)d_in[10];
  const float* W3r = (const float*)d_in[11];
  const float* We1 = (const float*)d_in[12];
  const float* be1 = (const float*)d_in[13];
  const float* We2 = (const float*)d_in[14];
  const float* be2 = (const float*)d_in[15];
  const float* We3 = (const float*)d_in[16];
  const float* be3 = (const float*)d_in[17];
  float* out = (float*)d_out;

  const int N = in_sizes[0] / DIN;
  const int E = in_sizes[1] / 2;
  const int* src = eidx;
  const int* dst = eidx + E;

  // ---- workspace carve-out ----
  char* base = (char*)d_ws;
  size_t off = 0;
  auto alloc = [&](size_t bytes) -> char* {
    char* p = base + off;
    off = (off + bytes + 255) & ~(size_t)255;
    return p;
  };
  float* cnt   = (float*)alloc((size_t)N * 4);
  float* inv   = (float*)alloc((size_t)N * 4);
  float* sum6  = (float*)alloc((size_t)N * DIN * 4);
  float* sum   = (float*)alloc((size_t)N * HDIM * 4);
  float* h1    = (float*)alloc((size_t)N * HDIM * 4);
  float* h2    = (float*)alloc((size_t)N * HDIM * 4);
  __bf16* h1bf  = (__bf16*)alloc((size_t)N * HDIM * 2);
  __bf16* h2bf  = (__bf16*)alloc((size_t)N * HDIM * 2);
  __bf16* h3bf  = (__bf16*)alloc((size_t)N * HDIM * 2);
  __bf16* mbf   = (__bf16*)alloc((size_t)N * HDIM * 2);
  __bf16* W2lbf = (__bf16*)alloc(HDIM * HDIM * 2);
  __bf16* W2rbf = (__bf16*)alloc(HDIM * HDIM * 2);
  __bf16* W3lbf = (__bf16*)alloc(HDIM * HDIM * 2);
  __bf16* W3rbf = (__bf16*)alloc(HDIM * HDIM * 2);
  __bf16* We1a  = (__bf16*)alloc(HDIM * HDIM * 2);
  __bf16* We1b  = (__bf16*)alloc(HDIM * HDIM * 2);
  __bf16* We2bf = (__bf16*)alloc(64 * HDIM * 2);
  (void)ws_size; (void)n_in; (void)out_size;

  const int T = 256;
  auto blocks = [](long n, int t) { return (unsigned)((n + t - 1) / t); };
  const long NH = (long)N * HDIM;
  const long EC = (long)E * (HDIM / 4);   // aggH: one thread per 4 features

  // weight conversions (cheap, independent)
  cvt_bf16<<<blocks(HDIM * HDIM, T), T, 0, stream>>>(W2l, W2lbf, HDIM * HDIM);
  cvt_bf16<<<blocks(HDIM * HDIM, T), T, 0, stream>>>(W2r, W2rbf, HDIM * HDIM);
  cvt_bf16<<<blocks(HDIM * HDIM, T), T, 0, stream>>>(W3l, W3lbf, HDIM * HDIM);
  cvt_bf16<<<blocks(HDIM * HDIM, T), T, 0, stream>>>(W3r, W3rbf, HDIM * HDIM);
  cvt_bf16<<<blocks(64 * HDIM, T), T, 0, stream>>>(We2, We2bf, 64 * HDIM);
  split_we1<<<blocks(HDIM * HDIM, T), T, 0, stream>>>(We1, We1a, We1b);

  // degrees
  zero_f32<<<blocks(N, T), T, 0, stream>>>(cnt, N);
  deg_kernel<<<blocks(E, T), T, 0, stream>>>(dst, cnt, E);
  inv_kernel<<<blocks(N, T), T, 0, stream>>>(cnt, inv, N);

  // ---- layer 1 ----
  zero_f32<<<blocks((long)N * DIN, T), T, 0, stream>>>(sum6, (long)N * DIN);
  agg6_kernel<<<blocks(E, T), T, 0, stream>>>(src, dst, x, sum6, E);
  node1_kernel<<<N, HDIM, 0, stream>>>(x, sum6, inv, W1l, b1l, W1r, h1, h1bf, N);

  // ---- layer 2 ----
  zero_f32<<<blocks(NH, T), T, 0, stream>>>(sum, NH);
  aggH_kernel<<<blocks(EC, T), T, 0, stream>>>(src, dst, h1, sum, EC);
  meanbf_kernel<<<blocks(NH, T), T, 0, stream>>>(sum, inv, mbf, NH);
  sage_node_wmma<<<(N + 15) / 16, 256, 0, stream>>>(mbf, h1bf, W2lbf, b2l, W2rbf,
                                                    nullptr, h2, h2bf, N);

  // ---- layer 3 (residual: h3 = relu(sage) + h2) ----
  zero_f32<<<blocks(NH, T), T, 0, stream>>>(sum, NH);
  aggH_kernel<<<blocks(EC, T), T, 0, stream>>>(src, dst, h2, sum, EC);
  meanbf_kernel<<<blocks(NH, T), T, 0, stream>>>(sum, inv, mbf, NH);
  sage_node_wmma<<<(N + 15) / 16, 256, 0, stream>>>(mbf, h2bf, W3lbf, b3l, W3rbf,
                                                    h2, h1 /*scratch*/, h3bf, N);

  // ---- fused edge MLP ----
  edge_mlp_kernel<<<(E + EPB - 1) / EPB, 256, 0, stream>>>(
      src, dst, edge_attr, h3bf, We1a, We1b, We1, be1, We2bf, be2, We3, be3, out, E);
}
